// WindowAttentionV2_55783035240623
// MI455X (gfx1250) — matrix operand order repaired and checked by reading, not compiled, not run on hardware
//
#include <hip/hip_runtime.h>
#include <hip/hip_bf16.h>

typedef __attribute__((ext_vector_type(16))) _Float16 v16h;
typedef __attribute__((ext_vector_type(8)))  float    v8f;

// ---------------------------------------------------------------------------
// WMMA helpers (CDNA5 gfx1250, wave32). D = A(16x32 f16) * B(32x16 f16) + C(f32)
// ---------------------------------------------------------------------------
__device__ __forceinline__ v8f wmma16(v16h a, v16h b, v8f c) {
    return __builtin_amdgcn_wmma_f32_16x16x32_f16(
        /*neg_a=*/false, a, /*neg_b=*/false, b,
        /*c_mod=*/(short)0, c, /*reuse_a=*/false, /*reuse_b=*/false);
}

// A fragment: 16x32 f16. Lane m / m+16 hold row m; element e -> K = e + 16g,
// g = lane>>4. One contiguous 16-half run -> 2x ds_load_b128.
__device__ __forceinline__ v16h load_a(const _Float16* p, int stride, int lane) {
    int g = (lane >> 4) & 1, m = lane & 15;
    v16h a;
#pragma unroll
    for (int e = 0; e < 16; ++e) {
        int k = e + 16 * g;
        a[e] = p[m * stride + k];
    }
    return a;
}

// B fragment from K-contiguous (transposed) storage [N][stride over K]:
// K = e + 16g, N = lane&15. One contiguous 16-half run -> 2x ds_load_b128.
__device__ __forceinline__ v16h load_bt(const _Float16* p, int stride, int lane) {
    int g = (lane >> 4) & 1, n = lane & 15;
    v16h b;
#pragma unroll
    for (int e = 0; e < 16; ++e) b[e] = p[n * stride + e + 16 * g];
    return b;
}

__device__ __forceinline__ _Float16 lo_h(uint32_t u) {
    return __builtin_bit_cast(_Float16, (unsigned short)(u & 0xffffu));
}
__device__ __forceinline__ _Float16 hi_h(uint32_t u) {
    return __builtin_bit_cast(_Float16, (unsigned short)(u >> 16));
}

// ---------------------------------------------------------------------------
// Kernel 1: continuous position bias MLP: t2[225][16]
// ---------------------------------------------------------------------------
__global__ __launch_bounds__(512) void cpb_mlp_kernel(
    const float* __restrict__ w1, const float* __restrict__ b1,
    const float* __restrict__ w2, float* __restrict__ t2) {
    __shared__ float hid[512];
    int p = blockIdx.x;            // 0..224  over (2W-1)^2, W=8
    int i = p / 15, j = p % 15;
    float c0 = (float)(i - 7) * (8.0f / 7.0f);
    float c1 = (float)(j - 7) * (8.0f / 7.0f);
    float t0 = copysignf(__log2f(fabsf(c0) + 1.0f) * (1.0f / 3.0f), c0);
    float t1 = copysignf(__log2f(fabsf(c1) + 1.0f) * (1.0f / 3.0f), c1);
    int h = threadIdx.x;
    float v = t0 * w1[h] + t1 * w1[512 + h] + b1[h];
    hid[h] = fmaxf(v, 0.0f);
    __syncthreads();
    if (h < 16) {
        float s = 0.0f;
        for (int q = 0; q < 512; ++q) s += hid[q] * w2[q * 16 + h];
        t2[p * 16 + h] = s;
    }
}

// ---------------------------------------------------------------------------
// Kernel 2: gather + 16*sigmoid -> bias[H=16][64][64]
// ---------------------------------------------------------------------------
__global__ __launch_bounds__(256) void cpb_bias_kernel(
    const float* __restrict__ t2, float* __restrict__ bias) {
    int gid = blockIdx.x * blockDim.x + threadIdx.x;   // 65536 total
    int h = gid >> 12;
    int rem = gid & 4095;
    int n = rem >> 6, m = rem & 63;
    int rn = n >> 3, cn = n & 7, rm = m >> 3, cm = m & 7;
    int idx = (rn - rm + 7) * 15 + (cn - cm + 7);
    float v = t2[idx * 16 + h];
    bias[gid] = 16.0f / (1.0f + __expf(-v));
}

// ---------------------------------------------------------------------------
// Kernel 3: QKV GEMM [131072 x 512] x [512 x 1536], f16 WMMA, fused
//           q/k row L2-normalization + logit-scale, emits q/k/v f16 [B,H,64,32]
// ---------------------------------------------------------------------------
__global__ __launch_bounds__(256) void qkv_gemm_kernel(
    const float* __restrict__ x, const float* __restrict__ w,
    const float* __restrict__ bqkv, const float* __restrict__ logit_scale,
    _Float16* __restrict__ qo, _Float16* __restrict__ ko, _Float16* __restrict__ vo) {
    const int ASTR = 40;          // 80 B rows, 16 B aligned
    const int BSTR = 40;          // transposed B: [n][k], 80 B rows
    alignas(16) __shared__ _Float16 As[128 * 40];
    alignas(16) __shared__ _Float16 BsT[64 * 40];
    int tid = threadIdx.x, lane = tid & 31, wid = tid >> 5;
    int waveM = wid >> 1, waveN = wid & 1;
    int rowBase = blockIdx.y * 128;
    int colBase = blockIdx.x * 64;

    v8f acc[2][2];
#pragma unroll
    for (int mt = 0; mt < 2; ++mt)
#pragma unroll
        for (int nt = 0; nt < 2; ++nt) acc[mt][nt] = (v8f)(0.0f);

    for (int kb = 0; kb < 512; kb += 32) {
        // ---- load phase: batch all global loads (no LDS stores in between) ----
        float4 xa[4];                                    // A: 128x32 f32, 1024 float4
#pragma unroll
        for (int i = 0; i < 4; ++i) {
            int fi = tid + i * 256;
            int r = fi >> 3, c4 = fi & 7;
            xa[i] = ((const float4*)(x + (size_t)(rowBase + r) * 512 + kb))[c4];
        }
        float4 wb[2];                                    // B: 32x64 f32, 512 float4
#pragma unroll
        for (int i = 0; i < 2; ++i) {
            int fi = tid + i * 256;
            int r = fi >> 4, c4 = fi & 15;
            wb[i] = ((const float4*)(w + (size_t)(kb + r) * 1536 + colBase))[c4];
        }
        if (kb + 32 < 512) {                             // prefetch next K tile
            __builtin_prefetch(x + (size_t)(rowBase + (tid >> 1)) * 512 + kb + 32, 0, 3);
            __builtin_prefetch(w + (size_t)(kb + 32 + (tid & 31)) * 1536 + colBase, 0, 3);
        }
        // ---- store phase ----
#pragma unroll
        for (int i = 0; i < 4; ++i) {                    // A: 4-half packs (b64)
            int fi = tid + i * 256;
            int r = fi >> 3, c4 = fi & 7;
            _Float16* dst = As + r * ASTR + c4 * 4;
            dst[0] = (_Float16)xa[i].x; dst[1] = (_Float16)xa[i].y;
            dst[2] = (_Float16)xa[i].z; dst[3] = (_Float16)xa[i].w;
        }
#pragma unroll
        for (int i = 0; i < 2; ++i) {                    // B: transposed, scalar b16
            int fi = tid + i * 256;
            int r = fi >> 4, c0 = (fi & 15) * 4;
            BsT[(c0 + 0) * BSTR + r] = (_Float16)wb[i].x;
            BsT[(c0 + 1) * BSTR + r] = (_Float16)wb[i].y;
            BsT[(c0 + 2) * BSTR + r] = (_Float16)wb[i].z;
            BsT[(c0 + 3) * BSTR + r] = (_Float16)wb[i].w;
        }
        __syncthreads();
        v16h af[2], bf[2];
#pragma unroll
        for (int mt = 0; mt < 2; ++mt)
            af[mt] = load_a(As + (waveM * 32 + mt * 16) * ASTR, ASTR, lane);
#pragma unroll
        for (int nt = 0; nt < 2; ++nt)
            bf[nt] = load_bt(BsT + (waveN * 32 + nt * 16) * BSTR, BSTR, lane);
#pragma unroll
        for (int mt = 0; mt < 2; ++mt)
#pragma unroll
            for (int nt = 0; nt < 2; ++nt)
                acc[mt][nt] = wmma16(af[mt], bf[nt], acc[mt][nt]);
        __syncthreads();
    }

    // Epilogue: wave's 32 cols == one (q|k|v, head) slice
    int g = (lane >> 4) & 1, ln = lane & 15;
    int wcol = colBase + waveN * 32;
    int which = wcol / 512;
    int h = (wcol % 512) / 32;
    float bb0 = bqkv[wcol + ln], bb1 = bqkv[wcol + 16 + ln];
    float scale = 1.0f;
    if (which == 0) scale = __expf(fminf(logit_scale[h], 4.6051702f)); // ln(100)

#pragma unroll
    for (int mt = 0; mt < 2; ++mt) {
#pragma unroll
        for (int r = 0; r < 8; ++r) {
            int grow = rowBase + waveM * 32 + mt * 16 + r + 8 * g;
            int b = grow >> 6, n = grow & 63;
            size_t ob = (((size_t)b * 16 + h) * 64 + n) * 32;
            float v0 = acc[mt][0][r] + bb0;
            float v1 = acc[mt][1][r] + bb1;
            if (which == 2) {
                vo[ob + ln]      = (_Float16)v0;
                vo[ob + 16 + ln] = (_Float16)v1;
            } else {
                float ss = v0 * v0 + v1 * v1;           // row sumsq over 32 dims
                for (int m2 = 1; m2 < 16; m2 <<= 1) ss += __shfl_xor(ss, m2, 32);
                float inv = 1.0f / fmaxf(sqrtf(ss), 1e-12f);
                float f = (which == 0) ? inv * scale : inv;
                _Float16* o = (which == 0) ? qo : ko;
                o[ob + ln]      = (_Float16)(v0 * f);
                o[ob + 16 + ln] = (_Float16)(v1 * f);
            }
        }
    }
}

// ---------------------------------------------------------------------------
// Kernel 4: attention per (b,h): S=q k^T (WMMA) + bias + mask, softmax,
//           O = P V (WMMA). 128 threads = 4 waves, wave w owns rows 16w..16w+15
// ---------------------------------------------------------------------------
__global__ __launch_bounds__(128) void attn_kernel(
    const _Float16* __restrict__ qg, const _Float16* __restrict__ kg,
    const _Float16* __restrict__ vg, const float* __restrict__ bias,
    const float* __restrict__ mask, _Float16* __restrict__ ao) {
    alignas(16) __shared__ _Float16 qs[64 * 32];     // [query][hd]
    alignas(16) __shared__ _Float16 ks[64 * 32];     // [key][hd]  (= B^T for S)
    const int VSTR = 72;
    alignas(16) __shared__ _Float16 vsT[32 * 72];    // [hd][key]  (= B^T for P@V)
    const int PSTR = 72;
    alignas(16) __shared__ _Float16 ps[64 * 72];     // [query][key] probabilities

    int h = blockIdx.x, b = blockIdx.y;
    int tid = threadIdx.x, lane = tid & 31, w = tid >> 5;
    size_t base = ((size_t)b * 16 + h) * 2048;

    // ---- load phase: 6x b128 loads batched ----
    const uint4* q16 = (const uint4*)(qg + base);    // 256 uint4 each
    const uint4* k16 = (const uint4*)(kg + base);
    const uint4* v16 = (const uint4*)(vg + base);
    uint4 qr[2], kr[2], vr[2];
#pragma unroll
    for (int i = 0; i < 2; ++i) {
        int idx = tid + i * 128;
        qr[i] = q16[idx]; kr[i] = k16[idx]; vr[i] = v16[idx];
    }
    // ---- store phase ----
#pragma unroll
    for (int i = 0; i < 2; ++i) {
        int idx = tid + i * 128;
        ((uint4*)qs)[idx] = qr[i];                   // b128 LDS stores
        ((uint4*)ks)[idx] = kr[i];
        int key = idx >> 2, d0 = (idx & 3) * 8;      // v: transpose while staging
        uint32_t u0 = vr[i].x, u1 = vr[i].y, u2 = vr[i].z, u3 = vr[i].w;
        vsT[(d0 + 0) * VSTR + key] = lo_h(u0);
        vsT[(d0 + 1) * VSTR + key] = hi_h(u0);
        vsT[(d0 + 2) * VSTR + key] = lo_h(u1);
        vsT[(d0 + 3) * VSTR + key] = hi_h(u1);
        vsT[(d0 + 4) * VSTR + key] = lo_h(u2);
        vsT[(d0 + 5) * VSTR + key] = hi_h(u2);
        vsT[(d0 + 6) * VSTR + key] = lo_h(u3);
        vsT[(d0 + 7) * VSTR + key] = hi_h(u3);
    }
    __syncthreads();

    // S = q @ k^T : 4 N-tiles, K=32 (one WMMA each)
    v8f s[4];
#pragma unroll
    for (int nt = 0; nt < 4; ++nt) s[nt] = (v8f)(0.0f);
    v16h aq = load_a(qs + w * 16 * 32, 32, lane);
#pragma unroll
    for (int nt = 0; nt < 4; ++nt) {
        v16h bk = load_bt(ks + nt * 16 * 32, 32, lane);
        s[nt] = wmma16(aq, bk, s[nt]);
    }

    int g = (lane >> 4) & 1, ln = lane & 15;
    const float* bh = bias + h * 4096;
    const float* mh = mask + (size_t)(b & 255) * 4096;
#pragma unroll
    for (int nt = 0; nt < 4; ++nt)
#pragma unroll
        for (int r = 0; r < 8; ++r) {
            int row = w * 16 + r + 8 * g, col = nt * 16 + ln;
            s[nt][r] += bh[row * 64 + col] + mh[row * 64 + col];
        }

    // softmax over 64 keys: per-row reduce across 4 tiles + 16-lane group
#pragma unroll
    for (int r = 0; r < 8; ++r) {
        float mx = fmaxf(fmaxf(s[0][r], s[1][r]), fmaxf(s[2][r], s[3][r]));
        for (int m2 = 1; m2 < 16; m2 <<= 1) mx = fmaxf(mx, __shfl_xor(mx, m2, 32));
        float e0 = __expf(s[0][r] - mx), e1 = __expf(s[1][r] - mx);
        float e2 = __expf(s[2][r] - mx), e3 = __expf(s[3][r] - mx);
        float sum = e0 + e1 + e2 + e3;
        for (int m2 = 1; m2 < 16; m2 <<= 1) sum += __shfl_xor(sum, m2, 32);
        float inv = 1.0f / sum;
        int row = w * 16 + r + 8 * g;
        ps[row * PSTR + 0  + ln] = (_Float16)(e0 * inv);
        ps[row * PSTR + 16 + ln] = (_Float16)(e1 * inv);
        ps[row * PSTR + 32 + ln] = (_Float16)(e2 * inv);
        ps[row * PSTR + 48 + ln] = (_Float16)(e3 * inv);
    }
    __syncthreads();

    // O = P @ V : K=64 keys in 2 chunks, N=32 dims in 2 tiles
    v8f o[2];
    o[0] = (v8f)(0.0f); o[1] = (v8f)(0.0f);
#pragma unroll
    for (int kc = 0; kc < 2; ++kc) {
        v16h ap = load_a(ps + w * 16 * PSTR + kc * 32, PSTR, lane);
#pragma unroll
        for (int nt = 0; nt < 2; ++nt) {
            v16h bv = load_bt(vsT + nt * 16 * VSTR + kc * 32, VSTR, lane);
            o[nt] = wmma16(ap, bv, o[nt]);
        }
    }
    // store to [b*64+n][h*32+d] f16 for proj GEMM
#pragma unroll
    for (int nt = 0; nt < 2; ++nt)
#pragma unroll
        for (int r = 0; r < 8; ++r) {
            int rowg = b * 64 + w * 16 + r + 8 * g;
            int colg = h * 32 + nt * 16 + ln;
            ao[(size_t)rowg * 512 + colg] = (_Float16)o[nt][r];
        }
}

// ---------------------------------------------------------------------------
// Kernel 5: projection GEMM [131072 x 512] x [512 x 512] + bias -> f32 out
// ---------------------------------------------------------------------------
__global__ __launch_bounds__(256) void proj_gemm_kernel(
    const _Float16* __restrict__ a, const float* __restrict__ w,
    const float* __restrict__ bp, float* __restrict__ out) {
    const int ASTR = 40, BSTR = 40;
    alignas(16) __shared__ _Float16 As[128 * 40];
    alignas(16) __shared__ _Float16 BsT[64 * 40];
    int tid = threadIdx.x, lane = tid & 31, wid = tid >> 5;
    int waveM = wid >> 1, waveN = wid & 1;
    int rowBase = blockIdx.y * 128;
    int colBase = blockIdx.x * 64;

    v8f acc[2][2];
#pragma unroll
    for (int mt = 0; mt < 2; ++mt)
#pragma unroll
        for (int nt = 0; nt < 2; ++nt) acc[mt][nt] = (v8f)(0.0f);

    for (int kb = 0; kb < 512; kb += 32) {
        // ---- load phase ----
        uint4 aa[2];                                     // A f16: 512 uint4 (8 halves)
#pragma unroll
        for (int i = 0; i < 2; ++i) {
            int fi = tid + i * 256;
            int r = fi >> 2, c8 = fi & 3;
            aa[i] = ((const uint4*)(a + (size_t)(rowBase + r) * 512 + kb))[c8];
        }
        float4 wb[2];                                    // B f32: 512 float4
#pragma unroll
        for (int i = 0; i < 2; ++i) {
            int fi = tid + i * 256;
            int r = fi >> 4, c4 = fi & 15;
            wb[i] = ((const float4*)(w + (size_t)(kb + r) * 512 + colBase))[c4];
        }
        if (kb + 32 < 512) {
            __builtin_prefetch(a + (size_t)(rowBase + (tid >> 1)) * 512 + kb + 32, 0, 3);
            __builtin_prefetch(w + (size_t)(kb + 32 + (tid & 31)) * 512 + colBase, 0, 3);
        }
        // ---- store phase ----
#pragma unroll
        for (int i = 0; i < 2; ++i) {                    // A: b128 LDS stores
            int fi = tid + i * 256;
            int r = fi >> 2, c8 = fi & 3;
            *(uint4*)(As + r * ASTR + c8 * 8) = aa[i];
        }
#pragma unroll
        for (int i = 0; i < 2; ++i) {                    // B: transposed, scalar b16
            int fi = tid + i * 256;
            int r = fi >> 4, c0 = (fi & 15) * 4;
            BsT[(c0 + 0) * BSTR + r] = (_Float16)wb[i].x;
            BsT[(c0 + 1) * BSTR + r] = (_Float16)wb[i].y;
            BsT[(c0 + 2) * BSTR + r] = (_Float16)wb[i].z;
            BsT[(c0 + 3) * BSTR + r] = (_Float16)wb[i].w;
        }
        __syncthreads();
        v16h af[2], bf[2];
#pragma unroll
        for (int mt = 0; mt < 2; ++mt)
            af[mt] = load_a(As + (waveM * 32 + mt * 16) * ASTR, ASTR, lane);
#pragma unroll
        for (int nt = 0; nt < 2; ++nt)
            bf[nt] = load_bt(BsT + (waveN * 32 + nt * 16) * BSTR, BSTR, lane);
#pragma unroll
        for (int mt = 0; mt < 2; ++mt)
#pragma unroll
            for (int nt = 0; nt < 2; ++nt)
                acc[mt][nt] = wmma16(af[mt], bf[nt], acc[mt][nt]);
        __syncthreads();
    }

    int g = (lane >> 4) & 1, ln = lane & 15;
#pragma unroll
    for (int mt = 0; mt < 2; ++mt)
#pragma unroll
        for (int nt = 0; nt < 2; ++nt) {
            int col = colBase + waveN * 32 + nt * 16 + ln;
            float bb = bp[col];
#pragma unroll
            for (int r = 0; r < 8; ++r) {
                int grow = rowBase + waveM * 32 + mt * 16 + r + 8 * g;
                out[(size_t)grow * 512 + col] = acc[mt][nt][r] + bb;
            }
        }
}

// ---------------------------------------------------------------------------
extern "C" void kernel_launch(void* const* d_in, const int* in_sizes, int n_in,
                              void* d_out, int out_size, void* d_ws, size_t ws_size,
                              hipStream_t stream) {
    (void)in_sizes; (void)n_in; (void)out_size; (void)ws_size;
    const float* x           = (const float*)d_in[0];
    const float* mask        = (const float*)d_in[1];
    const float* qkv_w       = (const float*)d_in[2];
    const float* qkv_b       = (const float*)d_in[3];
    const float* logit_scale = (const float*)d_in[4];
    const float* cpb_w1      = (const float*)d_in[5];
    const float* cpb_b1      = (const float*)d_in[6];
    const float* cpb_w2      = (const float*)d_in[7];
    const float* proj_w      = (const float*)d_in[8];
    const float* proj_b      = (const float*)d_in[9];
    float* out = (float*)d_out;

    char* ws = (char*)d_ws;
    const size_t QB = (size_t)2048 * 16 * 64 * 32 * sizeof(_Float16); // 134 MB
    _Float16* qf = (_Float16*)(ws);
    _Float16* kf = (_Float16*)(ws + QB);
    _Float16* vf = (_Float16*)(ws + 2 * QB);
    _Float16* ao = (_Float16*)(ws + 3 * QB);
    float* t2    = (float*)(ws + 4 * QB);
    float* bias  = (float*)(ws + 4 * QB + 16384);

    cpb_mlp_kernel<<<225, 512, 0, stream>>>(cpb_w1, cpb_b1, cpb_w2, t2);
    cpb_bias_kernel<<<256, 256, 0, stream>>>(t2, bias);
    qkv_gemm_kernel<<<dim3(24, 1024), 256, 0, stream>>>(x, qkv_w, qkv_b, logit_scale,
                                                        qf, kf, vf);
    attn_kernel<<<dim3(16, 2048), 128, 0, stream>>>(qf, kf, vf, bias, mask, ao);
    proj_gemm_kernel<<<dim3(8, 1024), 256, 0, stream>>>(ao, proj_w, proj_b, out);
}
